// ImprovedLinkingPredictor_7808250544872
// MI455X (gfx1250) — compile-verified, matrix operand order checked by reading, not run on hardware
//
#include <hip/hip_runtime.h>
#include <hip/hip_bf16.h>

typedef __attribute__((ext_vector_type(16))) _Float16 v16h;
typedef __attribute__((ext_vector_type(8)))  _Float16 v8h;
typedef __attribute__((ext_vector_type(8)))  float    v8f;
typedef unsigned int uint4v __attribute__((ext_vector_type(4)));
typedef int          int8v  __attribute__((ext_vector_type(8)));
typedef int          int4v  __attribute__((ext_vector_type(4)));

#define NTOK 768
#define DMODEL 256
#define STAGE_BYTES 20480           // 256 rows * (16 data + 4 pad) dwords
#define STAGE_STRIDE_H 40           // halfs per staged row (16B-aligned, bank-rotated)

__device__ __forceinline__ v16h cat16(v8h a, v8h b) {
  v16h r;
#pragma unroll
  for (int x = 0; x < 8; x++) { r[x] = a[x]; r[x + 8] = b[x]; }
  return r;
}

// Issue a TDM load of a 32-half-wide K-slice (256 rows) of a row-major f16
// weight matrix (row stride 256 halfs) into LDS with 4-dword row padding.
__device__ __forceinline__ void tdm_stage_slice(const _Float16* gsrc, unsigned int lds_off) {
  unsigned long long ga = (unsigned long long)(uintptr_t)gsrc;
  uint4v g0;
  g0[0] = 1u;                                   // count=1, user-mode descriptor
  g0[1] = lds_off;                              // lds_addr (bytes)
  g0[2] = (unsigned int)(ga & 0xFFFFFFFFu);     // global_addr[31:0]
  g0[3] = (unsigned int)((ga >> 32) & 0x1FFFFFFu) | (2u << 30);  // addr[56:32] | type=2
  int8v g1;
  g1[0] = (1 << 16)      // data_size = 2 bytes
        | (1 << 20)      // pad_enable
        | (3 << 22)      // pad_interval: 16 dwords
        | (3 << 25);     // pad_amount: 4 dwords
  g1[1] = (int)(32u << 16);     // tensor_dim0 = 32 (tile-relative width)
  g1[2] = (int)(256u << 16);    // tensor_dim1 = 256 rows
  g1[3] = (int)(32u << 16);     // tile_dim0 = 32
  g1[4] = 256;                  // tile_dim1 = 256 (tile_dim2 = 0)
  g1[5] = 256;                  // tensor_dim0_stride = 256 (low 32)
  g1[6] = 0;
  g1[7] = 0;
  int4v z4 = {0, 0, 0, 0};
  int8v z8 = {0, 0, 0, 0, 0, 0, 0, 0};
  __builtin_amdgcn_tensor_load_to_lds(g0, g1, z4, z4, z8, 0);
}

// ---------------------------------------------------------------------------
// Generic small GEMM: out[m,n] = sum_k X[m,k] * W[n*ldw + off + k] + bias[n]
// ---------------------------------------------------------------------------
__global__ void gemm_bias_k(const float* __restrict__ X, const float* __restrict__ W,
                            const float* __restrict__ bias, float* __restrict__ out,
                            int M, int N, int K, int ldw, int off) {
  int idx = blockIdx.x * 256 + threadIdx.x;
  if (idx >= M * N) return;
  int m = idx / N, n = idx % N;
  const float* x = X + m * K;
  const float* w = W + n * ldw + off;
  float s = bias ? bias[n] : 0.0f;
  for (int k = 0; k < K; k++) s = fmaf(x[k], w[k], s);
  out[idx] = s;
}

// ---------------------------------------------------------------------------
// MHA core: one block per (query i, head h). qkv row layout [q(256)|k(256)|v(256)]
// ---------------------------------------------------------------------------
__global__ void attn_k(const float* __restrict__ qkv, float* __restrict__ o) {
  int i = blockIdx.x, h = blockIdx.y, tid = threadIdx.x;
  __shared__ float sc[NTOK];
  __shared__ float red[256];
  __shared__ float part[8][32];
  const float* q = qkv + i * 768 + h * 32;
  for (int j = tid; j < NTOK; j += 256) {
    const float* kk = qkv + j * 768 + 256 + h * 32;
    float s = 0.0f;
#pragma unroll
    for (int d = 0; d < 32; d++) s = fmaf(q[d], kk[d], s);
    sc[j] = s * 0.17677669529663687f;  // 1/sqrt(32)
  }
  __syncthreads();
  float m = -1e30f;
  for (int j = tid; j < NTOK; j += 256) m = fmaxf(m, sc[j]);
  red[tid] = m; __syncthreads();
  for (int s = 128; s > 0; s >>= 1) { if (tid < s) red[tid] = fmaxf(red[tid], red[tid + s]); __syncthreads(); }
  float mx = red[0]; __syncthreads();
  float ps = 0.0f;
  for (int j = tid; j < NTOK; j += 256) { float e = __expf(sc[j] - mx); sc[j] = e; ps += e; }
  red[tid] = ps; __syncthreads();
  for (int s = 128; s > 0; s >>= 1) { if (tid < s) red[tid] += red[tid + s]; __syncthreads(); }
  float inv = 1.0f / red[0];
  __syncthreads();
  int d = tid & 31, sl = tid >> 5;
  float a = 0.0f;
  for (int j = sl * 96; j < sl * 96 + 96; j++) a = fmaf(sc[j], qkv[j * 768 + 512 + h * 32 + d], a);
  part[sl][d] = a; __syncthreads();
  if (tid < 32) {
    float s = 0.0f;
#pragma unroll
    for (int x = 0; x < 8; x++) s += part[x][tid];
    o[i * DMODEL + h * 32 + tid] = s * inv;
  }
}

// ---------------------------------------------------------------------------
// f32 -> f16 pack with strided source (also used to slice W1c out of w1)
// ---------------------------------------------------------------------------
__global__ void pack_f16_k(const float* __restrict__ src, _Float16* __restrict__ dst,
                           int rows, int cols, int stride, int off) {
  int idx = blockIdx.x * 256 + threadIdx.x;
  if (idx >= rows * cols) return;
  int r = idx / cols, c = idx % cols;
  dst[idx] = (_Float16)src[r * stride + off + c];
}

// ---------------------------------------------------------------------------
// Main fused pair-MLP kernel. 4 wave32 per block; wave = one (i, 16-j tile).
// Layer-1 weights are TDM-staged into LDS (double buffered), overlapped with
// WMMA, then the stage memory is reused as the activation transpose buffer.
// ---------------------------------------------------------------------------
#define WMMA16(A, B, C) __builtin_amdgcn_wmma_f32_16x16x32_f16(false, (A), false, (B), (short)0, (C), false, false)

__global__ __launch_bounds__(128)
void pair_mlp_k(const _Float16* __restrict__ attH, const _Float16* __restrict__ w1cH,
                const _Float16* __restrict__ w2H, const _Float16* __restrict__ w3H,
                const _Float16* __restrict__ w4H,
                const float* __restrict__ A_pre, const float* __restrict__ B_pre,
                const float* __restrict__ boxes, const float* __restrict__ w1,
                const float* __restrict__ ln_g, const float* __restrict__ ln_b,
                const float* __restrict__ b2, const float* __restrict__ b3,
                const float* __restrict__ b4, const float* __restrict__ w5,
                const float* __restrict__ b5, float* __restrict__ out) {
  // 2 x 20KB TDM stage buffers; later overlaid by 4 x 16x264-half act buffers.
  __shared__ __align__(128) char pool[2 * STAGE_BYTES];

  const int tid  = threadIdx.x;
  const int lane = tid & 31;
  const int wave = tid >> 5;
  const int row  = lane & 15;          // A-frag row / B-frag col / C-frag col within tile
  const int hi   = lane >> 4;          // lane half: selects K sub-chunk & C row half
  const int i    = blockIdx.y * 4 + wave;
  const int jbase = blockIdx.x * 16;
  const int jl   = jbase + row;        // this lane's A-matrix row -> pair (i, jl)

  const unsigned int poolOff = (unsigned int)(uintptr_t)(&pool[0]);  // LDS byte offset

  // ===================== Layer 1: h1 = relu(P @ W1c^T + pre) ================
  v8f acc[16];
#pragma unroll
  for (int t = 0; t < 16; t++)
#pragma unroll
    for (int e = 0; e < 8; e++) acc[t][e] = 0.0f;

  const _Float16* ai = attH + i * DMODEL;
  const _Float16* aj = attH + jl * DMODEL;

  if (wave == 0) tdm_stage_slice(w1cH, poolOff);          // prefetch slice 0

  for (int ks = 0; ks < 8; ks++) {
    if (wave == 0) __builtin_amdgcn_s_wait_tensorcnt(0);  // slice ks resident
    __syncthreads();                                      // ready + prev buffer free
    if (wave == 0 && ks + 1 < 8)
      tdm_stage_slice(w1cH + (ks + 1) * 32, poolOff + (unsigned)((ks + 1) & 1) * STAGE_BYTES);

    const int kb = ks * 32 + hi * 8;
    v8h p0 = (*(const v8h*)(ai + kb))      * (*(const v8h*)(aj + kb));
    v8h p1 = (*(const v8h*)(ai + kb + 16)) * (*(const v8h*)(aj + kb + 16));
    v16h A = cat16(p0, p1);
    const _Float16* sb = (const _Float16*)(pool + (ks & 1) * STAGE_BYTES);
#pragma unroll
    for (int t = 0; t < 16; t++) {
      const _Float16* wr = sb + (t * 16 + row) * STAGE_STRIDE_H + hi * 8;
      v16h B = cat16(*(const v8h*)(wr), *(const v8h*)(wr + 16));
      acc[t] = WMMA16(A, B, acc[t]);
    }
  }
  __syncthreads();   // all waves done with stage memory -> safe to overlay hbuf

  _Float16* hbw = (_Float16*)pool + wave * (16 * 264);   // per-wave 16x264 tile

  // epilogue: + A_pre[i] + B_pre[j] + W1s*spatial (+b1 folded in A_pre), relu
  const float bix = boxes[i * 4 + 0], biy = boxes[i * 4 + 1];
  float xd[8], yd[8];
#pragma unroll
  for (int r = 0; r < 8; r++) {
    int j = jbase + hi * 8 + r;
    xd[r] = fabsf(bix - boxes[j * 4 + 0]);
    yd[r] = biy - boxes[j * 4 + 1];
  }
#pragma unroll
  for (int t = 0; t < 16; t++) {
    const int n = t * 16 + row;
    const float apre = A_pre[i * 256 + n];
    const float sx = w1[n * 771 + 768], sy = w1[n * 771 + 769], sz = w1[n * 771 + 770];
#pragma unroll
    for (int r = 0; r < 8; r++) {
      int j = jbase + hi * 8 + r;
      float v = acc[t][r] + apre + B_pre[j * 256 + n]
              + sx * xd[r] + sy * fabsf(yd[r]) + sz * yd[r];
      acc[t][r] = fmaxf(v, 0.0f);
    }
  }

  // ===================== LayerNorm over 256 features =======================
  float mean[8], rstd[8];
#pragma unroll
  for (int r = 0; r < 8; r++) {
    float s1 = 0.0f, s2 = 0.0f;
#pragma unroll
    for (int t = 0; t < 16; t++) { float v = acc[t][r]; s1 += v; s2 += v * v; }
#pragma unroll
    for (int m = 1; m < 16; m <<= 1) { s1 += __shfl_xor(s1, m, 32); s2 += __shfl_xor(s2, m, 32); }
    float mu  = s1 * (1.0f / 256.0f);
    float var = s2 * (1.0f / 256.0f) - mu * mu;
    mean[r] = mu;
    rstd[r] = rsqrtf(var + 1e-5f);
  }
#pragma unroll
  for (int t = 0; t < 16; t++) {
    const int n = t * 16 + row;
    const float g = ln_g[n], bb = ln_b[n];
#pragma unroll
    for (int r = 0; r < 8; r++)
      acc[t][r] = (acc[t][r] - mean[r]) * rstd[r] * g + bb;
  }
  // C-layout -> row-major f16 in LDS (per-wave tile)
#pragma unroll
  for (int t = 0; t < 16; t++) {
    const int n = t * 16 + row;
#pragma unroll
    for (int r = 0; r < 8; r++) hbw[(hi * 8 + r) * 264 + n] = (_Float16)acc[t][r];
  }
  __syncthreads();

  // ===================== Layer 2: 256 -> 128 ===============================
  v8f a2[8];
#pragma unroll
  for (int t = 0; t < 8; t++)
#pragma unroll
    for (int e = 0; e < 8; e++) a2[t][e] = 0.0f;
  {
    const _Float16* hrow = hbw + row * 264;
    for (int ks = 0; ks < 8; ks++) {
      const int kb = ks * 32 + hi * 8;
      v16h A = cat16(*(const v8h*)(hrow + kb), *(const v8h*)(hrow + kb + 16));
#pragma unroll
      for (int t = 0; t < 8; t++) {
        const _Float16* wr = w2H + (t * 16 + row) * 256;
        v16h B = cat16(*(const v8h*)(wr + kb), *(const v8h*)(wr + kb + 16));
        a2[t] = WMMA16(A, B, a2[t]);
      }
    }
  }
  __syncthreads();
#pragma unroll
  for (int t = 0; t < 8; t++) {
    const int n = t * 16 + row;
    const float bb = b2[n];
#pragma unroll
    for (int r = 0; r < 8; r++)
      hbw[(hi * 8 + r) * 264 + n] = (_Float16)fmaxf(a2[t][r] + bb, 0.0f);
  }
  __syncthreads();

  // ===================== Layer 3: 128 -> 64 ================================
  v8f a3[4];
#pragma unroll
  for (int t = 0; t < 4; t++)
#pragma unroll
    for (int e = 0; e < 8; e++) a3[t][e] = 0.0f;
  {
    const _Float16* hrow = hbw + row * 264;
    for (int ks = 0; ks < 4; ks++) {
      const int kb = ks * 32 + hi * 8;
      v16h A = cat16(*(const v8h*)(hrow + kb), *(const v8h*)(hrow + kb + 16));
#pragma unroll
      for (int t = 0; t < 4; t++) {
        const _Float16* wr = w3H + (t * 16 + row) * 128;
        v16h B = cat16(*(const v8h*)(wr + kb), *(const v8h*)(wr + kb + 16));
        a3[t] = WMMA16(A, B, a3[t]);
      }
    }
  }
  __syncthreads();
#pragma unroll
  for (int t = 0; t < 4; t++) {
    const int n = t * 16 + row;
    const float bb = b3[n];
#pragma unroll
    for (int r = 0; r < 8; r++)
      hbw[(hi * 8 + r) * 264 + n] = (_Float16)fmaxf(a3[t][r] + bb, 0.0f);
  }
  __syncthreads();

  // ===================== Layer 4: 64 -> 32 =================================
  v8f a4[2];
#pragma unroll
  for (int t = 0; t < 2; t++)
#pragma unroll
    for (int e = 0; e < 8; e++) a4[t][e] = 0.0f;
  {
    const _Float16* hrow = hbw + row * 264;
#pragma unroll
    for (int ks = 0; ks < 2; ks++) {
      const int kb = ks * 32 + hi * 8;
      v16h A = cat16(*(const v8h*)(hrow + kb), *(const v8h*)(hrow + kb + 16));
#pragma unroll
      for (int t = 0; t < 2; t++) {
        const _Float16* wr = w4H + (t * 16 + row) * 64;
        v16h B = cat16(*(const v8h*)(wr + kb), *(const v8h*)(wr + kb + 16));
        a4[t] = WMMA16(A, B, a4[t]);
      }
    }
  }
  float h4[2][8];
#pragma unroll
  for (int t = 0; t < 2; t++) {
    const int n = t * 16 + row;
    const float bb = b4[n];
#pragma unroll
    for (int r = 0; r < 8; r++) h4[t][r] = fmaxf(a4[t][r] + bb, 0.0f);
  }

  // ===================== Layer 5: 32 -> 1 + diag mask ======================
  const float wa = w5[row], wb = w5[16 + row];
  const float b5v = b5[0];
#pragma unroll
  for (int r = 0; r < 8; r++) {
    float s = h4[0][r] * wa + h4[1][r] * wb;
#pragma unroll
    for (int m = 1; m < 16; m <<= 1) s += __shfl_xor(s, m, 32);
    s += b5v;
    const int mm = hi * 8 + r;
    const int j = jbase + mm;
    if (row == r) out[i * NTOK + j] = (i == j) ? -1.0e9f : s;
  }
}

// ---------------------------------------------------------------------------
extern "C" void kernel_launch(void* const* d_in, const int* in_sizes, int n_in,
                              void* d_out, int out_size, void* d_ws, size_t ws_size,
                              hipStream_t stream) {
  const float* features = (const float*)d_in[0];
  const float* boxes    = (const float*)d_in[1];
  const float* in_w     = (const float*)d_in[2];
  const float* in_b     = (const float*)d_in[3];
  const float* out_w    = (const float*)d_in[4];
  const float* out_b    = (const float*)d_in[5];
  const float* w1       = (const float*)d_in[6];
  const float* b1       = (const float*)d_in[7];
  const float* ln_g     = (const float*)d_in[8];
  const float* ln_b     = (const float*)d_in[9];
  const float* w2       = (const float*)d_in[10];
  const float* b2       = (const float*)d_in[11];
  const float* w3       = (const float*)d_in[12];
  const float* b3       = (const float*)d_in[13];
  const float* w4       = (const float*)d_in[14];
  const float* b4       = (const float*)d_in[15];
  const float* w5       = (const float*)d_in[16];
  const float* b5       = (const float*)d_in[17];
  float* out = (float*)d_out;

  // workspace layout (all chunks 16B-aligned)
  float* qkv    = (float*)d_ws;                 // 768*768
  float* o_att  = qkv   + 768 * 768;            // 768*256
  float* att    = o_att + 768 * 256;            // 768*256
  float* A_pre  = att   + 768 * 256;            // 768*256  (includes b1)
  float* B_pre  = A_pre + 768 * 256;            // 768*256
  _Float16* attH = (_Float16*)(B_pre + 768 * 256);   // 768*256
  _Float16* w1cH = attH + 768 * 256;            // 256*256
  _Float16* w2H  = w1cH + 256 * 256;            // 128*256
  _Float16* w3H  = w2H + 128 * 256;             // 64*128
  _Float16* w4H  = w3H + 64 * 128;              // 32*64

  // 1) qkv = features @ in_proj_w.T + in_proj_b   (768 x 768)
  gemm_bias_k<<<(768 * 768 + 255) / 256, 256, 0, stream>>>(features, in_w, in_b, qkv, 768, 768, 256, 256, 0);
  // 2) attention per (i, head)
  attn_k<<<dim3(768, 8), 256, 0, stream>>>(qkv, o_att);
  // 3) out projection
  gemm_bias_k<<<(768 * 256 + 255) / 256, 256, 0, stream>>>(o_att, out_w, out_b, att, 768, 256, 256, 256, 0);
  // 4) per-row precomputes: A_pre = att @ W1a.T + b1 ; B_pre = att @ W1b.T
  gemm_bias_k<<<(768 * 256 + 255) / 256, 256, 0, stream>>>(att, w1, b1, A_pre, 768, 256, 256, 771, 0);
  gemm_bias_k<<<(768 * 256 + 255) / 256, 256, 0, stream>>>(att, w1, nullptr, B_pre, 768, 256, 256, 771, 256);
  // 5) f16 packs
  pack_f16_k<<<(768 * 256 + 255) / 256, 256, 0, stream>>>(att, attH, 768, 256, 256, 0);
  pack_f16_k<<<(256 * 256 + 255) / 256, 256, 0, stream>>>(w1, w1cH, 256, 256, 771, 512);
  pack_f16_k<<<(128 * 256 + 255) / 256, 256, 0, stream>>>(w2, w2H, 128, 256, 256, 0);
  pack_f16_k<<<(64 * 128 + 255) / 256, 256, 0, stream>>>(w3, w3H, 64, 128, 128, 0);
  pack_f16_k<<<(32 * 64 + 255) / 256, 256, 0, stream>>>(w4, w4H, 32, 64, 64, 0);
  // 6) fused pair MLP: 48 j-tiles x 192 i-blocks (4 waves = 4 i's per block)
  pair_mlp_k<<<dim3(48, 192), 128, 0, stream>>>(attH, w1cH, w2H, w3H, w4H,
                                                A_pre, B_pre, boxes, w1,
                                                ln_g, ln_b, b2, b3, b4, w5, b5, out);
}
